// Inter_Correlation_Block_44178033607255
// MI455X (gfx1250) — compile-verified, matrix operand order checked by reading, not run on hardware
//
#include <hip/hip_runtime.h>

typedef __bf16 bf16_t;
typedef __attribute__((ext_vector_type(16))) __bf16 v16bf;
typedef __attribute__((ext_vector_type(8)))  float  v8f;

#define BN_EPS   1e-5f
#define BACKCAST 168
#define KPAD     192
#define HH1      512
#define HH2      64

// ---------------- weight pre-conversion ----------------
// W0: [512][168] f32 -> [512][192] bf16 zero-padded
__global__ __launch_bounds__(256) void k_pad_w0(const float* __restrict__ W0,
                                                bf16_t* __restrict__ W0b) {
  int i = blockIdx.x * 256 + threadIdx.x;
  if (i >= HH1 * KPAD) return;
  int n = i / KPAD, k = i - n * KPAD;
  W0b[i] = (bf16_t)(k < BACKCAST ? W0[n * BACKCAST + k] : 0.f);
}
// straight f32 -> bf16 copy (W1, gW0, gW1)
__global__ __launch_bounds__(256) void k_cvt(const float* __restrict__ src,
                                             bf16_t* __restrict__ dst, int total) {
  int i = blockIdx.x * 256 + threadIdx.x;
  if (i < total) dst[i] = (bf16_t)src[i];
}

// ---------------- degree / dinv ----------------
__global__ __launch_bounds__(256) void k_deg_init(float* __restrict__ deg, int N) {
  int i = blockIdx.x * 256 + threadIdx.x;
  if (i < N) deg[i] = 1.0f;               // self-loop weight
}
__global__ __launch_bounds__(256) void k_deg_edges(const int* __restrict__ dst,
                                                   const float* __restrict__ w,
                                                   float* __restrict__ deg, int E) {
  int e = blockIdx.x * 256 + threadIdx.x;
  if (e < E) atomicAdd(&deg[dst[e]], w[e]);
}
__global__ __launch_bounds__(256) void k_deg_inv(float* __restrict__ deg, int N) {
  int i = blockIdx.x * 256 + threadIdx.x;
  if (i < N) { float d = deg[i]; deg[i] = d > 0.f ? rsqrtf(d) : 0.f; }
}

// ---------------- fused MLP: x@W0^T -> relu/bn -> @W1^T -> relu/bn -> bf16 ----------------
// 32 rows per block; layer0: 8 waves x (2 m-tiles x 4 n-tiles); layer1: one tile/wave.
__global__ __launch_bounds__(256) void k_mlp(
    const float*  __restrict__ x,
    const bf16_t* __restrict__ W0b, const float* __restrict__ b0,
    const float* __restrict__ g0, const float* __restrict__ be0,
    const float* __restrict__ rm0, const float* __restrict__ rv0,
    const bf16_t* __restrict__ W1b, const float* __restrict__ b1,
    const float* __restrict__ g1, const float* __restrict__ be1,
    const float* __restrict__ rm1, const float* __restrict__ rv1,
    bf16_t* __restrict__ hout, int N)
{
  __shared__ bf16_t sx[32][KPAD];       // x tile (bf16), K padded to 192
  __shared__ bf16_t sh[32][520];        // layer-0 activation tile, skewed stride

  const int tid = threadIdx.x;
  const int m0  = blockIdx.x * 32;

  for (int i = tid; i < 32 * KPAD; i += 256) {
    int r = i / KPAD, c = i - (i / KPAD) * KPAD;
    float v = 0.f;
    if (c < BACKCAST && (m0 + r) < N) v = x[(size_t)(m0 + r) * BACKCAST + c];
    sx[r][c] = (bf16_t)v;
  }
  __syncthreads();

  const int wave = tid >> 5;
  const int lane = tid & 31;
  const int lh   = lane >> 4;     // lane half (K phase)
  const int lr   = lane & 15;     // row (A) / column (B,C)

  // ---- layer 0 ----
  v8f acc0[2][4] = {};
  for (int kc = 0; kc < 6; ++kc) {
    const int k0 = kc * 32;
    v16bf a[2];
#pragma unroll
    for (int mt = 0; mt < 2; ++mt)
#pragma unroll
      for (int j = 0; j < 8; ++j) {
        a[mt][j]     = sx[mt * 16 + lr][k0 + lh * 8 + j];
        a[mt][j + 8] = sx[mt * 16 + lr][k0 + 16 + lh * 8 + j];
      }
#pragma unroll
    for (int t = 0; t < 4; ++t) {
      const int n = (wave + t * 8) * 16 + lr;
      const bf16_t* wrow = W0b + (size_t)n * KPAD + k0;
      v16bf b;
#pragma unroll
      for (int j = 0; j < 8; ++j) {
        b[j]     = wrow[lh * 8 + j];
        b[j + 8] = wrow[16 + lh * 8 + j];
      }
      acc0[0][t] = __builtin_amdgcn_wmma_f32_16x16x32_bf16(
          false, a[0], false, b, (short)0, acc0[0][t], false, false);
      acc0[1][t] = __builtin_amdgcn_wmma_f32_16x16x32_bf16(
          false, a[1], false, b, (short)0, acc0[1][t], false, false);
    }
  }
  // epilogue 0: bias + relu + bn -> bf16 in LDS
#pragma unroll
  for (int t = 0; t < 4; ++t) {
    const int n = (wave + t * 8) * 16 + lr;
    const float bias = b0[n];
    const float sc = g0[n] * rsqrtf(rv0[n] + BN_EPS);
    const float sf = be0[n] - rm0[n] * sc;
#pragma unroll
    for (int mt = 0; mt < 2; ++mt)
#pragma unroll
      for (int v = 0; v < 8; ++v) {
        const int m = mt * 16 + v + 8 * lh;
        float z = acc0[mt][t][v] + bias;
        z = fmaxf(z, 0.f) * sc + sf;
        sh[m][n] = (bf16_t)z;
      }
  }
  __syncthreads();

  // ---- layer 1: 8 tiles (2m x 4n), one full-K tile per wave ----
  const int mt = wave >> 2;
  const int nt = wave & 3;
  const int n1 = nt * 16 + lr;
  v8f acc1 = {};
  const bf16_t* wrow1 = W1b + (size_t)n1 * HH1;
  for (int kc = 0; kc < 16; ++kc) {
    const int k0 = kc * 32;
    v16bf a, b;
#pragma unroll
    for (int j = 0; j < 8; ++j) {
      a[j]     = sh[mt * 16 + lr][k0 + lh * 8 + j];
      a[j + 8] = sh[mt * 16 + lr][k0 + 16 + lh * 8 + j];
      b[j]     = wrow1[k0 + lh * 8 + j];
      b[j + 8] = wrow1[k0 + 16 + lh * 8 + j];
    }
    acc1 = __builtin_amdgcn_wmma_f32_16x16x32_bf16(
        false, a, false, b, (short)0, acc1, false, false);
  }
  {
    const float bias = b1[n1];
    const float sc = g1[n1] * rsqrtf(rv1[n1] + BN_EPS);
    const float sf = be1[n1] - rm1[n1] * sc;
#pragma unroll
    for (int v = 0; v < 8; ++v) {
      const int m = m0 + mt * 16 + v + 8 * lh;
      float z = acc1[v] + bias;
      z = fmaxf(z, 0.f) * sc + sf;
      if (m < N) hout[(size_t)m * HH2 + n1] = (bf16_t)z;
    }
  }
}

// ---------------- GCN transform: t = h @ gW^T ; acc = dinv^2 * t (self loop) ----------------
// gW (bf16, 8KB) staged into LDS with async-to-LDS copies (ASYNCcnt).
__global__ __launch_bounds__(256) void k_gcn_xform(
    const bf16_t* __restrict__ hin, const bf16_t* __restrict__ gWb,
    const float* __restrict__ dinv,
    float* __restrict__ tbuf, float* __restrict__ acc, int N)
{
  __shared__ bf16_t sW[HH2][HH2];   // 8 KB

  const int tid = threadIdx.x;
  const int wave = tid >> 5, lane = tid & 31;
  const int lh = lane >> 4, lr = lane & 15;
  const int mbase = blockIdx.x * 128 + wave * 16;

  // async stage gW into LDS: 512 x 16B, 2 chunks per thread
  {
    unsigned ldsbase = (unsigned)(uintptr_t)(&sW[0][0]);
#pragma unroll
    for (int c = 0; c < 2; ++c) {
      int byteoff = (tid + c * 256) * 16;
      unsigned long long ga = (unsigned long long)(uintptr_t)((const char*)gWb + byteoff);
      unsigned la = ldsbase + (unsigned)byteoff;
      asm volatile("global_load_async_to_lds_b128 %0, %1, off"
                   :: "v"(la), "v"(ga) : "memory");
    }
    asm volatile("s_wait_asynccnt 0x0" ::: "memory");
  }
  __syncthreads();

  v8f c[4] = {};
  for (int kc = 0; kc < 2; ++kc) {
    const int k0 = kc * 32;
    const int row = mbase + lr;
    const bool ok = row < N;
    const bf16_t* arow = hin + (size_t)(ok ? row : 0) * HH2;
    v16bf a;
#pragma unroll
    for (int j = 0; j < 8; ++j) {
      a[j]     = ok ? arow[k0 + lh * 8 + j]      : (bf16_t)0.f;
      a[j + 8] = ok ? arow[k0 + 16 + lh * 8 + j] : (bf16_t)0.f;
    }
#pragma unroll
    for (int t = 0; t < 4; ++t) {
      const int n = t * 16 + lr;
      v16bf b;
#pragma unroll
      for (int j = 0; j < 8; ++j) {
        b[j]     = sW[n][k0 + lh * 8 + j];
        b[j + 8] = sW[n][k0 + 16 + lh * 8 + j];
      }
      c[t] = __builtin_amdgcn_wmma_f32_16x16x32_bf16(
          false, a, false, b, (short)0, c[t], false, false);
    }
  }

  float dv[8];
#pragma unroll
  for (int v = 0; v < 8; ++v) {
    const int m = mbase + v + 8 * lh;
    dv[v] = (m < N) ? dinv[m] : 0.f;
  }
#pragma unroll
  for (int t = 0; t < 4; ++t) {
    const int n = t * 16 + lr;
#pragma unroll
    for (int v = 0; v < 8; ++v) {
      const int m = mbase + v + 8 * lh;
      if (m < N) {
        const float tv = c[t][v];
        tbuf[(size_t)m * HH2 + n] = tv;
        acc[(size_t)m * HH2 + n]  = dv[v] * dv[v] * tv;  // self-loop contribution
      }
    }
  }
}

// ---------------- edge scatter: acc[dst] += dinv[src]*w*dinv[dst] * t[src] ----------------
__global__ __launch_bounds__(256) void k_edge_scatter(
    const int* __restrict__ src, const int* __restrict__ dst,
    const float* __restrict__ w, const float* __restrict__ dinv,
    const float* __restrict__ tbuf, float* __restrict__ acc, int E)
{
  const int e = blockIdx.x * 8 + (threadIdx.x >> 5);
  const int lane = threadIdx.x & 31;
  if (e >= E) return;
  const int s = src[e], d = dst[e];
  const float nrm = dinv[s] * w[e] * dinv[d];
  const float* trow = tbuf + (size_t)s * HH2;
  float* arow = acc + (size_t)d * HH2;
  atomicAdd(&arow[lane],      nrm * trow[lane]);
  atomicAdd(&arow[lane + 32], nrm * trow[lane + 32]);
}

// ---------------- finalize: bias + relu + bn -> bf16 (next layer) or f32 (output) ----------------
__global__ __launch_bounds__(256) void k_finalize(
    const float* __restrict__ acc, const float* __restrict__ bias,
    const float* __restrict__ g, const float* __restrict__ be,
    const float* __restrict__ rm, const float* __restrict__ rv,
    bf16_t* __restrict__ outb, float* __restrict__ outf, int total, int to_f32)
{
  const int i = blockIdx.x * 256 + threadIdx.x;
  if (i >= total) return;
  const int n = i & 63;
  float z = acc[i] + bias[n];
  const float sc = g[n] * rsqrtf(rv[n] + BN_EPS);
  z = fmaxf(z, 0.f) * sc + (be[n] - rm[n] * sc);
  if (to_f32) outf[i] = z;
  else        outb[i] = (bf16_t)z;
}

extern "C" void kernel_launch(void* const* d_in, const int* in_sizes, int n_in,
                              void* d_out, int out_size, void* d_ws, size_t ws_size,
                              hipStream_t stream)
{
  const float* x    = (const float*)d_in[0];
  const int*   ei   = (const int*)  d_in[1];
  const float* ew   = (const float*)d_in[2];
  const float* W0   = (const float*)d_in[3];
  const float* b0   = (const float*)d_in[4];
  const float* bn0g = (const float*)d_in[5];
  const float* bn0b = (const float*)d_in[6];
  const float* bn0m = (const float*)d_in[7];
  const float* bn0v = (const float*)d_in[8];
  const float* W1   = (const float*)d_in[9];
  const float* b1   = (const float*)d_in[10];
  const float* bn1g = (const float*)d_in[11];
  const float* bn1b = (const float*)d_in[12];
  const float* bn1m = (const float*)d_in[13];
  const float* bn1v = (const float*)d_in[14];
  const float* gW0  = (const float*)d_in[15];
  const float* gb0  = (const float*)d_in[16];
  const float* p0g  = (const float*)d_in[17];
  const float* p0b  = (const float*)d_in[18];
  const float* p0m  = (const float*)d_in[19];
  const float* p0v  = (const float*)d_in[20];
  const float* gW1  = (const float*)d_in[21];
  const float* gb1  = (const float*)d_in[22];
  const float* p1g  = (const float*)d_in[23];
  const float* p1b  = (const float*)d_in[24];
  const float* p1m  = (const float*)d_in[25];
  const float* p1v  = (const float*)d_in[26];

  const int N = in_sizes[0] / BACKCAST;
  const int E = in_sizes[1] / 2;
  const int* src = ei;
  const int* dst = ei + E;

  char* ws = (char*)d_ws;
  size_t off = 0;
  auto al = [](size_t v) { return (v + 255) & ~(size_t)255; };
  float*  nrm  = (float*) (ws + off); off += al((size_t)N * 4);
  bf16_t* hbuf = (bf16_t*)(ws + off); off += al((size_t)N * HH2 * 2);
  float*  tbuf = (float*) (ws + off); off += al((size_t)N * HH2 * 4);
  float*  acc  = (float*) (ws + off); off += al((size_t)N * HH2 * 4);
  bf16_t* W0b  = (bf16_t*)(ws + off); off += al((size_t)HH1 * KPAD * 2);
  bf16_t* W1b  = (bf16_t*)(ws + off); off += al((size_t)HH2 * HH1 * 2);
  bf16_t* g0b  = (bf16_t*)(ws + off); off += al((size_t)HH2 * HH2 * 2);
  bf16_t* g1b  = (bf16_t*)(ws + off); off += al((size_t)HH2 * HH2 * 2);
  (void)ws_size; (void)n_in; (void)out_size;

  const int total = N * HH2;

  // weight conversions (bf16, W0 zero-padded to K=192)
  k_pad_w0<<<(HH1 * KPAD + 255) / 256, 256, 0, stream>>>(W0, W0b);
  k_cvt<<<(HH2 * HH1 + 255) / 256, 256, 0, stream>>>(W1, W1b, HH2 * HH1);
  k_cvt<<<(HH2 * HH2 + 255) / 256, 256, 0, stream>>>(gW0, g0b, HH2 * HH2);
  k_cvt<<<(HH2 * HH2 + 255) / 256, 256, 0, stream>>>(gW1, g1b, HH2 * HH2);

  // degrees -> dinv (shared by both GCN layers)
  k_deg_init <<<(N + 255) / 256, 256, 0, stream>>>(nrm, N);
  k_deg_edges<<<(E + 255) / 256, 256, 0, stream>>>(dst, ew, nrm, E);
  k_deg_inv  <<<(N + 255) / 256, 256, 0, stream>>>(nrm, N);

  // fused two-layer MLP (WMMA bf16) -> hbuf
  k_mlp<<<(N + 31) / 32, 256, 0, stream>>>(x, W0b, b0, bn0g, bn0b, bn0m, bn0v,
                                           W1b, b1, bn1g, bn1b, bn1m, bn1v, hbuf, N);

  // GCN layer 0
  k_gcn_xform  <<<(N + 127) / 128, 256, 0, stream>>>(hbuf, g0b, nrm, tbuf, acc, N);
  k_edge_scatter<<<(E + 7) / 8, 256, 0, stream>>>(src, dst, ew, nrm, tbuf, acc, E);
  k_finalize   <<<(total + 255) / 256, 256, 0, stream>>>(acc, gb0, p0g, p0b, p0m, p0v,
                                                         hbuf, nullptr, total, 0);
  // GCN layer 1
  k_gcn_xform  <<<(N + 127) / 128, 256, 0, stream>>>(hbuf, g1b, nrm, tbuf, acc, N);
  k_edge_scatter<<<(E + 7) / 8, 256, 0, stream>>>(src, dst, ew, nrm, tbuf, acc, E);
  k_finalize   <<<(total + 255) / 256, 256, 0, stream>>>(acc, gb1, p1g, p1b, p1m, p1v,
                                                         nullptr, (float*)d_out, total, 1);
}
